// SelfAttention_robotcar_56968446214803
// MI455X (gfx1250) — matrix operand order repaired and checked by reading, not compiled, not run on hardware
//
#include <hip/hip_runtime.h>
#include <hip/hip_bf16.h>

// ---------------------------------------------------------------------------
// MSDeformAttn (single level) for MI455X / gfx1250, wave32 + WMMA f16.
//   B=2, NQ=257*512=131584, D=128, HEADS=8, PTS=4, DH=16, grid H=257, W=512
// Pipeline:
//   k0/k1: prepack weights into WMMA B-fragment order (f16)
//   k2   : fused projection GEMM (N=224) -> f16 value planes + off/attn buffer
//          (weight fragments staged in LDS via async-to-LDS loads, inline asm)
//   k3   : softmax + bilinear sampling -> LDS -> output GEMM (+bias+2*query)
//          (W_out fragments async-staged into LDS, overlapped with gather)
// ---------------------------------------------------------------------------

#define BSZ    2
#define HGRID  257
#define WGRID  512
#define NQ     (HGRID * WGRID)      // 131584
#define DMODEL 128
#define HEADS  8
#define PTS    4
#define DH     16
#define NPROJ  224                  // 128 value + 64 off + 32 attn

typedef __attribute__((ext_vector_type(16))) _Float16 v16h;
typedef __attribute__((ext_vector_type(8)))  float    v8f;

union FragU { uint4 u[2]; v16h h; };
union PackU { _Float16 h[2]; unsigned int u; };

// ---------------------------------------------------------------------------
// CDNA5 async global->LDS copy (ASYNCcnt-tracked), via inline asm since the
// ROCm builtin's parameter types (__device__-AS vector pointers) are not
// spellable from HIP. VDST = per-lane LDS byte offset (low 32 bits of the
// generic LDS pointer), VADDR = 64-bit global address, no SADDR.
// ---------------------------------------------------------------------------
__device__ inline void async_b128(const void* gsrc, void* ldst) {
    unsigned int       loff  = (unsigned int)(unsigned long long)ldst;
    unsigned long long gaddr = (unsigned long long)gsrc;
    asm volatile("global_load_async_to_lds_b128 %0, %1, off"
                 :: "v"(loff), "v"(gaddr) : "memory");
}
__device__ inline void wait_async0() {
    asm volatile("s_wait_asynccnt 0" ::: "memory");
}

// ---------------------------------------------------------------------------
// Weight prepack: fragment f = ntile*4 + kchunk, 1024 bytes each.
// Within a fragment, lane n owns 16 halves (32B): half j ->
//   k   = kchunk*32 + (n>>4)*16 + j
//   col = ntile*16 + (n&15)
// Contiguous k order, not the ISA table order; the A fragments use the same order.
// ---------------------------------------------------------------------------
__global__ void pack_proj_weights(const float* __restrict__ Wv,
                                  const float* __restrict__ Wo,
                                  const float* __restrict__ Wa,
                                  _Float16* __restrict__ frag) {
    int t = blockIdx.x * blockDim.x + threadIdx.x;
    if (t >= DMODEL * NPROJ) return;
    int f    = t >> 9;            // 512 halves per fragment
    int idx  = t & 511;
    int lane = idx >> 4;
    int j    = idx & 15;
    int ntile = f >> 2;
    int kc    = f & 3;
    int k   = kc * 32 + ((lane >> 4) << 4) + j;
    int col = ntile * 16 + (lane & 15);
    float w;
    if (col < 128)      w = Wv[k * 128 + col];
    else if (col < 192) w = Wo[k * 64 + (col - 128)];
    else                w = Wa[k * 32 + (col - 192)];
    frag[(size_t)f * 512 + lane * 16 + j] = (_Float16)w;
}

__global__ void pack_out_weights(const float* __restrict__ Wout,
                                 _Float16* __restrict__ frag) {
    int t = blockIdx.x * blockDim.x + threadIdx.x;
    if (t >= DMODEL * DMODEL) return;
    int f    = t >> 9;
    int idx  = t & 511;
    int lane = idx >> 4;
    int j    = idx & 15;
    int ntile = f >> 2;
    int kc    = f & 3;
    int k   = kc * 32 + ((lane >> 4) << 4) + j;
    int col = ntile * 16 + (lane & 15);
    frag[(size_t)f * 512 + lane * 16 + j] = (_Float16)Wout[k * 128 + col];
}

// ---------------------------------------------------------------------------
// Projection GEMM: [128 rows/block] x [N=224], K=128, f16 WMMA, f32 accum.
// Weight fragments (56 KB) staged once per block in LDS (async on CDNA5).
// ---------------------------------------------------------------------------
__global__ __launch_bounds__(256) void proj_kernel(
        const float* __restrict__ query, const _Float16* __restrict__ fragP,
        const float* __restrict__ b_value, const float* __restrict__ b_off,
        const float* __restrict__ b_attn,
        _Float16* __restrict__ ws_value, float* __restrict__ ws_proj) {
    __shared__ unsigned int sfrag[NPROJ * DMODEL / 2];  // 57344 B, 3584 uint4

    // ---- stage all 56 B-fragments into LDS (async, ASYNCcnt) ----
    #pragma unroll
    for (int i = 0; i < 14; ++i) {
        const int chunk = threadIdx.x + i * 256;      // < 3584
        async_b128((const char*)fragP + (size_t)chunk * 16, sfrag + chunk * 4);
    }

    const int wave = threadIdx.x >> 5;
    const int lane = threadIdx.x & 31;
    const int lrow = lane & 15;
    const int lhi  = lane >> 4;
    const long rowg = (long)blockIdx.x * 128 + wave * 16;  // global row in B*NQ
    const int b  = (int)(rowg / NQ);
    const int q0 = (int)(rowg % NQ);

    // ---- A fragments: 4 K-chunks of query (fp32 -> f16), overlaps staging ----
    const float* qrow = query + (rowg + lrow) * DMODEL;
    v16h a[4];
    #pragma unroll
    for (int kc = 0; kc < 4; ++kc) {
        const int kb = kc * 32 + lhi * 16;
        float4 f0 = *(const float4*)(qrow + kb);
        float4 f1 = *(const float4*)(qrow + kb + 4);
        float4 f2 = *(const float4*)(qrow + kb + 8);
        float4 f3 = *(const float4*)(qrow + kb + 12);
        v16h t;
        t[0]=(_Float16)f0.x; t[1]=(_Float16)f0.y; t[2]=(_Float16)f0.z; t[3]=(_Float16)f0.w;
        t[4]=(_Float16)f1.x; t[5]=(_Float16)f1.y; t[6]=(_Float16)f1.z; t[7]=(_Float16)f1.w;
        t[8]=(_Float16)f2.x; t[9]=(_Float16)f2.y; t[10]=(_Float16)f2.z; t[11]=(_Float16)f2.w;
        t[12]=(_Float16)f3.x; t[13]=(_Float16)f3.y; t[14]=(_Float16)f3.z; t[15]=(_Float16)f3.w;
        a[kc] = t;
    }

    wait_async0();
    __syncthreads();

    // ---- 14 N-tiles: value (0..7), offsets (8..11), attn (12..13) ----
    for (int nt = 0; nt < 14; ++nt) {
        v8f c = {};
        #pragma unroll
        for (int kc = 0; kc < 4; ++kc) {
            const uint4* bp =
                (const uint4*)((const char*)sfrag + (size_t)(nt * 4 + kc) * 1024) + lane * 2;
            FragU B; B.u[0] = bp[0]; B.u[1] = bp[1];
            c = __builtin_amdgcn_wmma_f32_16x16x32_f16(
                    false, a[kc], false, B.h, (short)0, c, false, false);
        }
        const int col = nt * 16 + lrow;
        float bias;
        if (col < 128)      bias = b_value[col];
        else if (col < 192) bias = b_off[col - 128];
        else                bias = b_attn[col - 192];

        if (nt < 8) {
            // value -> f16 head-major plane [(b*8+head)][q][dh]
            _Float16* base = ws_value + (((long)(b * HEADS + nt) * NQ + q0) * DH) + lrow;
            #pragma unroll
            for (int r = 0; r < 8; ++r)
                base[(long)(r + lhi * 8) * DH] = (_Float16)(c[r] + bias);
        } else {
            // offsets+logits -> [b][q][96] fp32
            const int pc = (nt - 8) * 16 + lrow;  // 0..95
            #pragma unroll
            for (int r = 0; r < 8; ++r)
                ws_proj[((long)b * NQ + q0 + r + lhi * 8) * 96 + pc] = c[r] + bias;
        }
    }
}

// ---------------------------------------------------------------------------
// Sampling + output GEMM.
// Phase 0: async-stage W_out fragments (32 KB) into LDS (overlaps phase 1).
// Phase 1: (row, head) items: softmax over 4 pts, f16 bilinear gather of the
//          DH=16 channels, fp32 accumulate, pack f16 -> LDS [128][128].
// Phase 2: waves read A-fragments from LDS, WMMA against W_out fragments in
//          LDS, fuse bias + 2*query residual into the store.
// ---------------------------------------------------------------------------
__global__ __launch_bounds__(256) void samp_out_kernel(
        const float* __restrict__ query, const _Float16* __restrict__ ws_value,
        const float* __restrict__ ws_proj, const _Float16* __restrict__ fragO,
        const float* __restrict__ b_out, float* __restrict__ out) {
    __shared__ unsigned int smem[128 * 64 + 128 * 64];  // 32 KB sampled + 32 KB W_out
    unsigned int* sout  = smem;            // sampled tile, [row][64 uints]
    unsigned int* swout = smem + 128 * 64; // W_out fragments, 2048 uint4

    // ---- phase 0: stage W_out fragments (completes under phase 1) ----
    #pragma unroll
    for (int i = 0; i < 8; ++i) {
        const int chunk = threadIdx.x + i * 256;      // < 2048
        async_b128((const char*)fragO + (size_t)chunk * 16, swout + chunk * 4);
    }

    const long rowg0 = (long)blockIdx.x * 128;
    const int b  = (int)(rowg0 / NQ);
    const int q0 = (int)(rowg0 % NQ);

    // ---------------- Phase 1: gather ----------------
    #pragma unroll
    for (int iter = 0; iter < 4; ++iter) {
        const int it   = iter * 256 + threadIdx.x;
        const int row  = it >> 3;
        const int head = it & 7;
        const int q  = q0 + row;
        const int ry = q / WGRID;
        const int cx = q - ry * WGRID;

        const float* pr = ws_proj + ((long)b * NQ + q) * 96;
        float4 o0 = *(const float4*)(pr + head * 8);
        float4 o1 = *(const float4*)(pr + head * 8 + 4);
        float4 lg = *(const float4*)(pr + 64 + head * 4);

        // softmax over the 4 points
        float m  = fmaxf(fmaxf(lg.x, lg.y), fmaxf(lg.z, lg.w));
        float e0 = __expf(lg.x - m), e1 = __expf(lg.y - m);
        float e2 = __expf(lg.z - m), e3 = __expf(lg.w - m);
        float inv = 1.0f / (e0 + e1 + e2 + e3);
        float wp[4] = { e0 * inv, e1 * inv, e2 * inv, e3 * inv };
        float offx[4] = { o0.x, o0.z, o1.x, o1.z };
        float offy[4] = { o0.y, o0.w, o1.y, o1.w };

        // pixel coords: x = gx*512 - 0.5 + off_x (offset normalizer cancels)
        const float xbase = (float)cx * (512.0f / 511.0f) - 0.5f;
        const float ybase = (float)ry * (257.0f / 256.0f) - 0.5f;

        const _Float16* plane = ws_value + (long)(b * HEADS + head) * NQ * DH;
        float acc[DH];
        #pragma unroll
        for (int j = 0; j < DH; ++j) acc[j] = 0.0f;

        #pragma unroll
        for (int p = 0; p < PTS; ++p) {
            float x = xbase + offx[p];
            float y = ybase + offy[p];
            float x0f = floorf(x), y0f = floorf(y);
            int   x0 = (int)x0f,  y0 = (int)y0f;
            float wx1 = x - x0f, wx0 = 1.0f - wx1;
            float wy1 = y - y0f, wy0 = 1.0f - wy1;
            const int   xi[4] = { x0, x0 + 1, x0, x0 + 1 };
            const int   yi[4] = { y0, y0, y0 + 1, y0 + 1 };
            const float cw[4] = { wx0 * wy0, wx1 * wy0, wx0 * wy1, wx1 * wy1 };
            #pragma unroll
            for (int cn = 0; cn < 4; ++cn) {
                if (xi[cn] < 0 || xi[cn] >= WGRID || yi[cn] < 0 || yi[cn] >= HGRID)
                    continue;
                const float w = wp[p] * cw[cn];
                const uint4* cp =
                    (const uint4*)(plane + ((long)yi[cn] * WGRID + xi[cn]) * DH);
                FragU V; V.u[0] = cp[0]; V.u[1] = cp[1];
                #pragma unroll
                for (int j = 0; j < DH; ++j)
                    acc[j] += w * (float)V.h[j];
            }
        }

        unsigned int* dst = sout + row * 64 + head * 8;
        #pragma unroll
        for (int j = 0; j < 8; ++j) {
            PackU pk;
            pk.h[0] = (_Float16)acc[2 * j];
            pk.h[1] = (_Float16)acc[2 * j + 1];
            dst[j] = pk.u;
        }
    }

    wait_async0();
    __syncthreads();

    // ---------------- Phase 2: out = sampled @ W_out + b_out + 2*query ------
    const int wave = threadIdx.x >> 5;
    const int lane = threadIdx.x & 31;
    const int lrow = lane & 15;
    const int lhi  = lane >> 4;
    const int srow = wave * 16 + lrow;

    v16h a[4];
    #pragma unroll
    for (int kc = 0; kc < 4; ++kc) {
        const int kb = kc * 32 + lhi * 16;            // halves
        uint4 u0 = *(const uint4*)(sout + srow * 64 + (kb >> 1));
        uint4 u1 = *(const uint4*)(sout + srow * 64 + ((kb + 8) >> 1));
        FragU A; A.u[0] = u0; A.u[1] = u1;
        a[kc] = A.h;
    }

    for (int nt = 0; nt < 8; ++nt) {
        v8f c = {};
        #pragma unroll
        for (int kc = 0; kc < 4; ++kc) {
            const uint4* bp =
                (const uint4*)((const char*)swout + (size_t)(nt * 4 + kc) * 1024) + lane * 2;
            FragU B; B.u[0] = bp[0]; B.u[1] = bp[1];
            c = __builtin_amdgcn_wmma_f32_16x16x32_f16(
                    false, a[kc], false, B.h, (short)0, c, false, false);
        }
        const int col  = nt * 16 + lrow;
        const float bias = b_out[col];
        #pragma unroll
        for (int r = 0; r < 8; ++r) {
            const long gr = rowg0 + wave * 16 + r + lhi * 8;
            const float qv = query[gr * DMODEL + col];
            out[gr * DMODEL + col] = c[r] + bias + 2.0f * qv;
        }
    }
}

// ---------------------------------------------------------------------------
extern "C" void kernel_launch(void* const* d_in, const int* in_sizes, int n_in,
                              void* d_out, int out_size, void* d_ws, size_t ws_size,
                              hipStream_t stream) {
    const float* query   = (const float*)d_in[0];
    const float* W_value = (const float*)d_in[1];
    const float* b_value = (const float*)d_in[2];
    const float* W_off   = (const float*)d_in[3];
    const float* b_off   = (const float*)d_in[4];
    const float* W_attn  = (const float*)d_in[5];
    const float* b_attn  = (const float*)d_in[6];
    const float* W_out   = (const float*)d_in[7];
    const float* b_out   = (const float*)d_in[8];
    float* out = (float*)d_out;

    char* ws = (char*)d_ws;
    _Float16* fragP    = (_Float16*)ws;                       //  57344 B
    _Float16* fragO    = (_Float16*)(ws + 57344);             //  32768 B
    _Float16* ws_value = (_Float16*)(ws + 90112);             // 67371008 B (f16)
    float*    ws_proj  = (float*)(ws + 90112 +
                           (size_t)BSZ * HEADS * NQ * DH * sizeof(_Float16));
    // total ws required: ~169 MB

    pack_proj_weights<<<(DMODEL * NPROJ + 255) / 256, 256, 0, stream>>>(
        W_value, W_off, W_attn, fragP);
    pack_out_weights<<<(DMODEL * DMODEL + 255) / 256, 256, 0, stream>>>(
        W_out, fragO);

    const int nblocks = (BSZ * NQ) / 128;  // 2056
    proj_kernel<<<nblocks, 256, 0, stream>>>(
        query, fragP, b_value, b_off, b_attn, ws_value, ws_proj);
    samp_out_kernel<<<nblocks, 256, 0, stream>>>(
        query, ws_value, ws_proj, fragO, b_out, out);
}
